// Model_29119878266979
// MI455X (gfx1250) — compile-verified, matrix operand order checked by reading, not run on hardware
//
#include <hip/hip_runtime.h>
#include <math.h>

typedef __attribute__((ext_vector_type(16))) _Float16 v16h;
typedef __attribute__((ext_vector_type(8)))  float    v8f;
typedef __attribute__((ext_vector_type(4)))  int      v4i_t;
typedef _Float16 h16;

#define D_MODEL 512
#define HEAD_DIM 64
#define WIN_LEN 25

#define GF_RES   1
#define GF_GELU  2
#define GF_OUT16 4

#if defined(__gfx1250__) && __has_builtin(__builtin_amdgcn_global_load_async_to_lds_b128)
#define USE_ASYNC 1
#else
#define USE_ASYNC 0
#endif

__device__ __forceinline__ void async_copy16(const h16* g, h16* l) {
#if USE_ASYNC
  __builtin_amdgcn_global_load_async_to_lds_b128(
      (__attribute__((address_space(1))) v4i_t*)(unsigned long long)(uintptr_t)g,
      (__attribute__((address_space(3))) v4i_t*)(unsigned)(uintptr_t)l, 0, 0);
#else
  (void)g; (void)l;
#endif
}

__device__ __forceinline__ void wait_async0() {
#if USE_ASYNC
#if __has_builtin(__builtin_amdgcn_s_wait_asynccnt)
  __builtin_amdgcn_s_wait_asynccnt(0);
#else
  asm volatile("s_wait_asynccnt 0" ::: "memory");
#endif
#endif
}

__device__ __forceinline__ v16h ldfrag(const h16* p0, const h16* p1) {
  union { uint4 u[2]; v16h v; } t;
  t.u[0] = *(const uint4*)p0;
  t.u[1] = *(const uint4*)p1;
  return t.v;
}

__device__ __forceinline__ float gelu_t(float x) {
  return 0.5f * x * (1.0f + tanhf(0.7978845608028654f * (x + 0.044715f * x * x * x)));
}

// ---------------- GEMM: C[M,N](f32) = A[M,K](f16) @ B[K,N](f16) + bias (+res, gelu, f16-out)
// 128x64 block tile, BK=32, double-buffered LDS, async A staging when available.
#define BM 128
#define BN 64
#define BK 32
#define SA 40
#define SB 40

template <int FLAGS>
__global__ __launch_bounds__(256) void gemm_f16(
    const h16* __restrict__ A, const h16* __restrict__ Bw,
    const float* __restrict__ bias, const float* __restrict__ resid,
    float* __restrict__ Cf, h16* __restrict__ Ch,
    int M, int N, int K)
{
  __shared__ h16 As[2][BM * SA];
  __shared__ h16 Bs[2][BN * SB];
  const int t = threadIdx.x;
  const int lane = t & 31, wave = t >> 5;
  const int wm = wave >> 1, wn = wave & 1;
  const int ml = lane & 15, kh = lane >> 4;
  const int m0 = blockIdx.x * BM, n0 = blockIdx.y * BN;

  const int ra = t >> 1, ca = (t & 1) * 16;   // A staging: 16 halves per thread
  const int rb = t >> 3, cb = (t & 7) * 8;    // B staging: 8 halves per thread

  const h16* Arow = A + (size_t)(m0 + ra) * K + ca;

  v8f c[2][2] = {};

  // ---- stage tile 0 into buffer 0
  {
    uint4 bx = *(const uint4*)(Bw + (size_t)rb * N + (n0 + cb));
    const h16* hx = (const h16*)&bx;
#pragma unroll
    for (int e = 0; e < 8; ++e) Bs[0][(cb + e) * SB + rb] = hx[e];
#if USE_ASYNC
    async_copy16(Arow, &As[0][ra * SA + ca]);
    async_copy16(Arow + 8, &As[0][ra * SA + ca + 8]);
    wait_async0();
#else
    const uint4* sA = (const uint4*)Arow;
    uint4 x0 = sA[0], x1 = sA[1];
    *(uint4*)(&As[0][ra * SA + ca])     = x0;
    *(uint4*)(&As[0][ra * SA + ca + 8]) = x1;
#endif
  }
  __syncthreads();

  const int nk = K / BK;
  for (int kt = 0; kt < nk; ++kt) {
    const int cur = kt & 1, nxt = cur ^ 1;
    const bool have_next = (kt + 1 < nk);
    uint4 bx = {0, 0, 0, 0};
#if !USE_ASYNC
    uint4 ax0 = {0, 0, 0, 0}, ax1 = {0, 0, 0, 0};
#endif
    if (have_next) {
      const int k0 = (kt + 1) * BK;
      bx = *(const uint4*)(Bw + (size_t)(k0 + rb) * N + (n0 + cb));
#if USE_ASYNC
      async_copy16(Arow + k0, &As[nxt][ra * SA + ca]);
      async_copy16(Arow + k0 + 8, &As[nxt][ra * SA + ca + 8]);
#else
      const uint4* sA = (const uint4*)(Arow + k0);
      ax0 = sA[0]; ax1 = sA[1];
#endif
    }

    // compute on current buffer
    v16h af[2], bf[2];
#pragma unroll
    for (int i = 0; i < 2; ++i) {
      const h16* p = &As[cur][(wm * 32 + i * 16 + ml) * SA];
      af[i] = ldfrag(p + kh * 8, p + 16 + kh * 8);
    }
#pragma unroll
    for (int j = 0; j < 2; ++j) {
      const h16* p = &Bs[cur][(wn * 32 + j * 16 + ml) * SB];
      bf[j] = ldfrag(p + kh * 8, p + 16 + kh * 8);
    }
#pragma unroll
    for (int i = 0; i < 2; ++i)
#pragma unroll
      for (int j = 0; j < 2; ++j)
        c[i][j] = __builtin_amdgcn_wmma_f32_16x16x32_f16(
            false, af[i], false, bf[j], (short)0, c[i][j], false, false);

    if (have_next) {
      const h16* hx = (const h16*)&bx;
#pragma unroll
      for (int e = 0; e < 8; ++e) Bs[nxt][(cb + e) * SB + rb] = hx[e];
#if !USE_ASYNC
      *(uint4*)(&As[nxt][ra * SA + ca])     = ax0;
      *(uint4*)(&As[nxt][ra * SA + ca + 8]) = ax1;
#endif
      wait_async0();
    }
    __syncthreads();
  }

  // ---- epilogue (compile-time specialized)
#pragma unroll
  for (int j = 0; j < 2; ++j) {
    int n = n0 + wn * 32 + j * 16 + ml;
    float bv = bias[n];
#pragma unroll
    for (int i = 0; i < 2; ++i) {
#pragma unroll
      for (int r = 0; r < 8; ++r) {
        int m = m0 + wm * 32 + i * 16 + r + 8 * kh;
        float v = c[i][j][r] + bv;
        if (FLAGS & GF_RES)  v += resid[(size_t)m * N + n];
        if (FLAGS & GF_GELU) v = gelu_t(v);
        if (FLAGS & GF_OUT16) Ch[(size_t)m * N + n] = (h16)v;
        else                  Cf[(size_t)m * N + n] = v;
      }
    }
  }
}

// ---------------- log-sparse flash attention (head_dim=64, key tiles of 32)
#define QS 72
#define KSS 72
#define VSS 40
#define PSS 40
#define NEGBIG (-1e30f)

__global__ __launch_bounds__(128) void attn_sparse(
    const float* __restrict__ Q, const float* __restrict__ Kb,
    const float* __restrict__ V, float* __restrict__ O,
    int Lq, int Lk, int causal)
{
  __shared__ h16 Qs[64 * QS];
  __shared__ h16 Ksm[32 * KSS];
  __shared__ h16 Vsm[64 * VSS];
  __shared__ h16 Psm[4 * 16 * PSS];

  const int t = threadIdx.x;
  const int lane = t & 31, wave = t >> 5;
  const int ml = lane & 15, kh = lane >> 4;
  const int qt0 = blockIdx.x * 64;
  const int h = blockIdx.y;
  const int b = blockIdx.z;

  for (int idx = t; idx < 64 * 64; idx += 128) {
    int r = idx >> 6, cc = idx & 63;
    Qs[r * QS + cc] = (h16)Q[((size_t)(b * Lq + qt0 + r)) * D_MODEL + h * HEAD_DIM + cc];
  }
  __syncthreads();

  v16h aQ[2];
#pragma unroll
  for (int kc = 0; kc < 2; ++kc) {
    const h16* p = &Qs[(wave * 16 + ml) * QS + kc * 32];
    aQ[kc] = ldfrag(p + kh * 8, p + 16 + kh * 8);
  }

  v8f o[4] = {};
  float mrow[8], srow[8];
#pragma unroll
  for (int r = 0; r < 8; ++r) { mrow[r] = NEGBIG; srow[r] = 0.0f; }

  const float scale = 0.125f;
  const int nkt = Lk >> 5;

  for (int kt = 0; kt < nkt; ++kt) {
    const int kk = kt * 32;
    if (causal && kk > qt0 + 63) continue;
    // tile-level log-sparse activity test (uniform over block)
    int a0 = qt0, a1 = qt0 + 63, c0 = kk, c1 = kk + 31;
    int dmin = 0;
    if (c0 > a1) dmin = c0 - a1;
    else if (a0 > c1) dmin = a0 - c1;
    bool active = (dmin <= WIN_LEN);
    if (!active) {
      int d1 = a1 >= c0 ? a1 - c0 : c0 - a1;
      int d2 = a0 >= c1 ? a0 - c1 : c1 - a0;
      int dmax = d1 > d2 ? d1 : d2;
      int e = dmin - WIN_LEN;                // >= 1 here
      int p2 = 1 << (32 - __clz(e - 1));     // next pow2 >= e
      active = (p2 <= dmax - WIN_LEN);
    }
    if (!active) continue;

    __syncthreads();
    for (int idx = t; idx < 32 * 64; idx += 128) {
      int r = idx >> 6, cc = idx & 63;
      size_t row = (size_t)(b * Lk + kk + r);
      Ksm[r * KSS + cc] = (h16)Kb[row * D_MODEL + h * HEAD_DIM + cc];
      Vsm[cc * VSS + r] = (h16)V[row * D_MODEL + h * HEAD_DIM + cc];
    }
    __syncthreads();

    float sv[2][8];
#pragma unroll
    for (int j = 0; j < 2; ++j) {
      v8f sa = {};
      const h16* p = &Ksm[(j * 16 + ml) * KSS];
      v16h b0 = ldfrag(p + kh * 8, p + 16 + kh * 8);
      v16h b1 = ldfrag(p + 32 + kh * 8, p + 48 + kh * 8);
      sa = __builtin_amdgcn_wmma_f32_16x16x32_f16(false, aQ[0], false, b0, (short)0, sa, false, false);
      sa = __builtin_amdgcn_wmma_f32_16x16x32_f16(false, aQ[1], false, b1, (short)0, sa, false, false);
#pragma unroll
      for (int r = 0; r < 8; ++r) {
        int qpos = qt0 + wave * 16 + r + 8 * kh;
        int kpos = kk + j * 16 + ml;
        int dist = qpos >= kpos ? qpos - kpos : kpos - qpos;
        int dd = dist - WIN_LEN;
        bool ok = (dist <= WIN_LEN) || ((dd > 0) && ((dd & (dd - 1)) == 0));
        if (causal) ok = ok && (kpos <= qpos);
        sv[j][r] = ok ? sa[r] * scale : NEGBIG;
      }
    }

    float alp[8];
#pragma unroll
    for (int r = 0; r < 8; ++r) {
      float lm = fmaxf(sv[0][r], sv[1][r]);
      for (int off = 8; off > 0; off >>= 1) lm = fmaxf(lm, __shfl_xor(lm, off, 32));
      float mn = fmaxf(mrow[r], lm);
      float al = __expf(mrow[r] - mn);
      mrow[r] = mn;
      float ps = 0.0f;
#pragma unroll
      for (int j = 0; j < 2; ++j) {
        float val = sv[j][r];
        float pj = (val <= 0.5f * NEGBIG) ? 0.0f : __expf(val - mn);
        Psm[wave * 16 * PSS + (r + 8 * kh) * PSS + j * 16 + ml] = (h16)pj;
        ps += pj;
      }
      for (int off = 8; off > 0; off >>= 1) ps += __shfl_xor(ps, off, 32);
      srow[r] = srow[r] * al + ps;
      alp[r] = al;
    }
#pragma unroll
    for (int d = 0; d < 4; ++d)
#pragma unroll
      for (int r = 0; r < 8; ++r) o[d][r] *= alp[r];

    __syncthreads();
    const h16* pp = &Psm[wave * 16 * PSS + ml * PSS];
    v16h aP = ldfrag(pp + kh * 8, pp + 16 + kh * 8);
#pragma unroll
    for (int d = 0; d < 4; ++d) {
      const h16* vp = &Vsm[(d * 16 + ml) * VSS];
      v16h bV = ldfrag(vp + kh * 8, vp + 16 + kh * 8);
      o[d] = __builtin_amdgcn_wmma_f32_16x16x32_f16(false, aP, false, bV, (short)0, o[d], false, false);
    }
  }

#pragma unroll
  for (int d = 0; d < 4; ++d)
#pragma unroll
    for (int r = 0; r < 8; ++r) {
      int qpos = qt0 + wave * 16 + r + 8 * kh;
      O[((size_t)(b * Lq + qpos)) * D_MODEL + h * HEAD_DIM + d * 16 + ml] = o[d][r] / srow[r];
    }
}

// ---------------- small kernels
__global__ void embed_k(const float* __restrict__ x, const float* __restrict__ mark,
                        const float* __restrict__ convw, const float* __restrict__ markw,
                        float* __restrict__ out, int L, int Cin)
{
  int r = blockIdx.x;
  int d = threadIdx.x;
  int b = r / L, l = r % L;
  float acc = 0.0f;
  for (int k = 0; k < 3; ++k) {
    int ls = l + k - 1; ls = (ls + L) % L;
    const float* xs = x + ((size_t)(b * L + ls)) * Cin;
    for (int i = 0; i < Cin; ++i) acc += xs[i] * convw[d * Cin * 3 + i * 3 + k];
  }
  const float* ms = mark + ((size_t)r) * 4;
  for (int m = 0; m < 4; ++m) acc += ms[m] * markw[m * D_MODEL + d];
  int j = d >> 1;
  float dv = __expf(-(float)(2 * j) * (9.210340371976184f / 512.0f));
  float ang = (float)l * dv;
  acc += (d & 1) ? cosf(ang) : sinf(ang);
  out[(size_t)r * D_MODEL + d] = acc;
}

// im2col for QK_KER=4 causal conv: out[r, k*512+i] = x[b, l+k-3, i]; vectorized x4
__global__ void im2col4_k(const float* __restrict__ x, h16* __restrict__ out, int L)
{
  size_t idx = (size_t)blockIdx.x * 256 + threadIdx.x;   // over M*512 quads
  int c4 = (int)(idx & 511);
  size_t r = idx >> 9;
  int k = c4 >> 7;
  int i = (c4 & 127) * 4;
  int l = (int)(r % L);
  int bb = (int)(r / L);
  int ls = l + k - 3;
  float4 v = make_float4(0.f, 0.f, 0.f, 0.f);
  if (ls >= 0) v = *(const float4*)(x + ((size_t)(bb * L + ls)) * D_MODEL + i);
  union { h16 h[4]; uint2 u; } pk;
  pk.h[0] = (h16)v.x; pk.h[1] = (h16)v.y; pk.h[2] = (h16)v.z; pk.h[3] = (h16)v.w;
  *(uint2*)(out + (idx << 2)) = pk.u;
}

__global__ void cvt4_f32_f16(const float* __restrict__ in, h16* __restrict__ out, size_t n4)
{
  size_t idx = (size_t)blockIdx.x * 256 + threadIdx.x;
  if (idx >= n4) return;
  float4 v = *(const float4*)(in + (idx << 2));
  union { h16 h[4]; uint2 u; } pk;
  pk.h[0] = (h16)v.x; pk.h[1] = (h16)v.y; pk.h[2] = (h16)v.z; pk.h[3] = (h16)v.w;
  *(uint2*)(out + (idx << 2)) = pk.u;
}

__global__ void convT_qk(const float* __restrict__ in, h16* __restrict__ out)
{
  size_t idx = (size_t)blockIdx.x * 256 + threadIdx.x; // 2048*512
  int o = (int)(idx & 511);
  int row = (int)(idx >> 9);
  int kk = row >> 9, i = row & 511;
  out[idx] = (h16)in[(size_t)o * 2048 + i * 4 + kk];
}

__global__ __launch_bounds__(256) void layernorm_k(float* __restrict__ x,
    const float* __restrict__ g, const float* __restrict__ bta, int M)
{
  int lane = threadIdx.x & 31, wave = threadIdx.x >> 5;
  int row = blockIdx.x * 8 + wave;
  if (row >= M) return;
  float* xr = x + (size_t)row * D_MODEL;
  float v[16];
  float s = 0.0f;
#pragma unroll
  for (int i = 0; i < 16; ++i) { v[i] = xr[i * 32 + lane]; s += v[i]; }
  for (int off = 16; off > 0; off >>= 1) s += __shfl_xor(s, off, 32);
  float mean = s * (1.0f / 512.0f);
  float q = 0.0f;
#pragma unroll
  for (int i = 0; i < 16; ++i) { float d = v[i] - mean; q += d * d; }
  for (int off = 16; off > 0; off >>= 1) q += __shfl_xor(q, off, 32);
  float rs = rsqrtf(q * (1.0f / 512.0f) + 1e-5f);
#pragma unroll
  for (int i = 0; i < 16; ++i) {
    int cc = i * 32 + lane;
    xr[cc] = (v[i] - mean) * rs * g[cc] + bta[cc];
  }
}

__global__ void proj_k(const float* __restrict__ x, const float* __restrict__ pw,
                       const float* __restrict__ pb, float* __restrict__ out)
{
  int idx = blockIdx.x * 256 + threadIdx.x;
  if (idx >= 8 * 256 * 7) return;
  int c = idx % 7;
  int r = idx / 7;
  int b = r >> 8;
  int l = 512 + (r & 255);
  const float* xr = x + ((size_t)(b * 768 + l)) * D_MODEL;
  float acc = pb[c];
  for (int d = 0; d < D_MODEL; ++d) acc += xr[d] * pw[d * 7 + c];
  out[idx] = acc;
}

// ---------------- host orchestration
extern "C" void kernel_launch(void* const* d_in, const int* in_sizes, int n_in,
                              void* d_out, int out_size, void* d_ws, size_t ws_size,
                              hipStream_t stream)
{
  const float* in[80];
  for (int i = 0; i < n_in && i < 80; ++i) in[i] = (const float*)d_in[i];

  char* ws = (char*)d_ws;
  size_t cur = 0;
  auto alloc = [&](size_t bytes) -> void* {
    size_t o = (cur + 255) & ~(size_t)255;
    cur = o + bytes;
    return (void*)(ws + o);
  };

  const int attnBase[4] = {8, 24, 42, 50};   // enc0, enc1, dec-self, dec-cross
  const int ffnBase[3]  = {16, 32, 58};      // enc0, enc1, dec

  h16 *wqT[4], *wkT[4], *wvh[4], *woh[4];
  for (int a = 0; a < 4; ++a) {
    wqT[a] = (h16*)alloc((size_t)2048 * 512 * 2);
    wkT[a] = (h16*)alloc((size_t)2048 * 512 * 2);
    wvh[a] = (h16*)alloc((size_t)512 * 512 * 2);
    woh[a] = (h16*)alloc((size_t)512 * 512 * 2);
  }
  h16 *w1h[3], *w2h[3];
  for (int f = 0; f < 3; ++f) {
    w1h[f] = (h16*)alloc((size_t)512 * 2048 * 2);
    w2h[f] = (h16*)alloc((size_t)2048 * 512 * 2);
  }
  float* encx = (float*)alloc((size_t)8192 * 512 * 4);
  float* decx = (float*)alloc((size_t)6144 * 512 * 4);
  h16*   xf16 = (h16*)  alloc((size_t)8192 * 512 * 2);
  float* qb   = (float*)alloc((size_t)8192 * 512 * 4);
  float* kb   = (float*)alloc((size_t)8192 * 512 * 4);
  float* vb   = (float*)alloc((size_t)8192 * 512 * 4);
  float* ab   = (float*)alloc((size_t)8192 * 512 * 4);
  h16*   big0 = (h16*)  alloc((size_t)8192 * 2048 * 2);
  (void)ws_size; (void)in_sizes; (void)out_size;

  auto cvt = [&](const float* src, h16* dst, size_t n) {
    cvt4_f32_f16<<<(unsigned)(n / 1024), 256, 0, stream>>>(src, dst, n / 4);
  };

  for (int a = 0; a < 4; ++a) {
    convT_qk<<<(2048 * 512) / 256, 256, 0, stream>>>(in[attnBase[a] + 0], wqT[a]);
    convT_qk<<<(2048 * 512) / 256, 256, 0, stream>>>(in[attnBase[a] + 2], wkT[a]);
    cvt(in[attnBase[a] + 4], wvh[a], (size_t)512 * 512);
    cvt(in[attnBase[a] + 6], woh[a], (size_t)512 * 512);
  }
  for (int f = 0; f < 3; ++f) {
    cvt(in[ffnBase[f] + 0], w1h[f], (size_t)512 * 2048);
    cvt(in[ffnBase[f] + 2], w2h[f], (size_t)512 * 2048);
  }

  embed_k<<<8 * 1024, 512, 0, stream>>>(in[0], in[1], in[4], in[5], encx, 1024, 7);
  embed_k<<<8 * 768,  512, 0, stream>>>(in[2], in[3], in[6], in[7], decx, 768, 7);

  auto gemm = [&](const h16* A, const h16* W, const float* bias, const float* resid,
                  float* Cf, h16* Ch, int M, int N, int K, int flags) {
    dim3 g(M / 128, N / 64);
    switch (flags) {
      case GF_RES:
        gemm_f16<GF_RES><<<g, 256, 0, stream>>>(A, W, bias, resid, Cf, Ch, M, N, K); break;
      case GF_GELU | GF_OUT16:
        gemm_f16<GF_GELU | GF_OUT16><<<g, 256, 0, stream>>>(A, W, bias, resid, Cf, Ch, M, N, K); break;
      default:
        gemm_f16<0><<<g, 256, 0, stream>>>(A, W, bias, resid, Cf, Ch, M, N, K); break;
    }
  };

  // ---- encoder (2 layers) ----
  for (int li = 0; li < 2; ++li) {
    int base = 8 + 16 * li;
    const int M = 8192, L = 1024;
    im2col4_k<<<(M * 512) / 256, 256, 0, stream>>>(encx, big0, L);
    gemm(big0, wqT[li], in[base + 1], nullptr, qb, nullptr, M, 512, 2048, 0);
    gemm(big0, wkT[li], in[base + 3], nullptr, kb, nullptr, M, 512, 2048, 0);
    cvt(encx, xf16, (size_t)M * 512);
    gemm(xf16, wvh[li], in[base + 5], nullptr, vb, nullptr, M, 512, 512, 0);
    attn_sparse<<<dim3(L / 64, 8, 8), 128, 0, stream>>>(qb, kb, vb, ab, L, L, 0);
    cvt(ab, xf16, (size_t)M * 512);
    gemm(xf16, woh[li], in[base + 7], encx, encx, nullptr, M, 512, 512, GF_RES);
    layernorm_k<<<M / 8, 256, 0, stream>>>(encx, in[base + 12], in[base + 13], M);
    cvt(encx, xf16, (size_t)M * 512);
    gemm(xf16, w1h[li], in[base + 9], nullptr, nullptr, big0, M, 2048, 512, GF_GELU | GF_OUT16);
    gemm(big0, w2h[li], in[base + 11], encx, encx, nullptr, M, 512, 2048, GF_RES);
    layernorm_k<<<M / 8, 256, 0, stream>>>(encx, in[base + 14], in[base + 15], M);
  }
  layernorm_k<<<8192 / 8, 256, 0, stream>>>(encx, in[40], in[41], 8192);

  // ---- decoder (1 layer) ----
  {
    const int Md = 6144, Ld = 768, Me = 8192, Le = 1024;
    // self-attention (causal)
    im2col4_k<<<(Md * 512) / 256, 256, 0, stream>>>(decx, big0, Ld);
    gemm(big0, wqT[2], in[43], nullptr, qb, nullptr, Md, 512, 2048, 0);
    gemm(big0, wkT[2], in[45], nullptr, kb, nullptr, Md, 512, 2048, 0);
    cvt(decx, xf16, (size_t)Md * 512);
    gemm(xf16, wvh[2], in[47], nullptr, vb, nullptr, Md, 512, 512, 0);
    attn_sparse<<<dim3(Ld / 64, 8, 8), 128, 0, stream>>>(qb, kb, vb, ab, Ld, Ld, 1);
    cvt(ab, xf16, (size_t)Md * 512);
    gemm(xf16, woh[2], in[49], decx, decx, nullptr, Md, 512, 512, GF_RES);
    layernorm_k<<<Md / 8, 256, 0, stream>>>(decx, in[62], in[63], Md);
    // cross-attention
    im2col4_k<<<(Md * 512) / 256, 256, 0, stream>>>(decx, big0, Ld);
    gemm(big0, wqT[3], in[51], nullptr, qb, nullptr, Md, 512, 2048, 0);
    im2col4_k<<<(Me * 512) / 256, 256, 0, stream>>>(encx, big0, Le);
    gemm(big0, wkT[3], in[53], nullptr, kb, nullptr, Me, 512, 2048, 0);
    cvt(encx, xf16, (size_t)Me * 512);
    gemm(xf16, wvh[3], in[55], nullptr, vb, nullptr, Me, 512, 512, 0);
    attn_sparse<<<dim3(Ld / 64, 8, 8), 128, 0, stream>>>(qb, kb, vb, ab, Ld, Le, 0);
    cvt(ab, xf16, (size_t)Md * 512);
    gemm(xf16, woh[3], in[57], decx, decx, nullptr, Md, 512, 512, GF_RES);
    layernorm_k<<<Md / 8, 256, 0, stream>>>(decx, in[64], in[65], Md);
    // ffn
    cvt(decx, xf16, (size_t)Md * 512);
    gemm(xf16, w1h[2], in[59], nullptr, nullptr, big0, Md, 2048, 512, GF_GELU | GF_OUT16);
    gemm(big0, w2h[2], in[61], decx, decx, nullptr, Md, 512, 2048, GF_RES);
    layernorm_k<<<Md / 8, 256, 0, stream>>>(decx, in[66], in[67], Md);
    layernorm_k<<<Md / 8, 256, 0, stream>>>(decx, in[68], in[69], Md);
  }

  proj_k<<<(8 * 256 * 7) / 256, 256, 0, stream>>>(decx, in[70], in[71], (float*)d_out);
}
